// LSTMTrajectoryPredictor_1451698946193
// MI455X (gfx1250) — matrix operand order repaired and checked by reading, not compiled, not run on hardware
//
#include <hip/hip_runtime.h>
#include <cstdint>
#include <cstddef>

// ---------------------------------------------------------------------------
// LSTM seq2seq trajectory predictor for gfx1250 (MI455X).
// B=16384, T=100, I=6, H=128, G=4H=512, OUT_LEN=20, O=2.
//
// bf16 WMMA (v_wmma_f32_16x16x32_bf16) for all gate matmuls, f32 cell state
// in registers (WMMA C/D layout), bf16 h-state in LDS, weights pre-converted
// to bf16 in d_ws (L2-resident, ~0.93 MB).
// Block = 256 threads = 8 waves = 4 batch tiles of 16 rows; each wave-pair
// shares one batch tile and splits the 512 gate outputs (64 hidden cols/wave).
//
// Round-4: kt-outer / q-inner WMMA order -> four independent accumulator
// chains (gates i,f,g,o) interleaved, so B-fragment L2 loads for the next
// gates overlap the current WMMA instead of an s_wait_loadcnt 0 per WMMA.
// ---------------------------------------------------------------------------

typedef __attribute__((ext_vector_type(16))) __bf16 v16bf;
typedef __attribute__((ext_vector_type(8)))  float  v8f;

#define T_STEPS 100
#define I_DIM   6
#define H_DIM   128
#define OUTLEN  20
#define O_DIM   2
#define LSTR    136   // LDS row stride in halves (128 + 8 pad -> conflict-free b128 reads)

__device__ __forceinline__ unsigned short f2bf(float f) {
  unsigned u = __float_as_uint(f);
  u += 0x7FFFu + ((u >> 16) & 1u);   // round-to-nearest-even
  return (unsigned short)(u >> 16);
}
__device__ __forceinline__ float bf2f(unsigned short h) {
  return __uint_as_float(((unsigned)h) << 16);
}
__device__ __forceinline__ float sigm(float x)  { return 1.f / (1.f + __expf(-x)); }
__device__ __forceinline__ float tanh_(float x) { return 2.f / (1.f + __expf(-2.f * x)) - 1.f; }

// B fragment (K=32, N=16): 32 contiguous bytes at a precomputed lane base +
// compile-time element offset. Base = W + (ctbase*16 + n)*cols + kh*16.
__device__ __forceinline__ v16bf loadB(const unsigned short* __restrict__ base, int eoff) {
  return *reinterpret_cast<const v16bf*>(base + eoff);
}

// A fragment (M=16, K=32) from LDS h buffer via lane base hA = hb + m*LSTR + kh*16.
// eoff = kt*32 (compile-time) -> ds_load_b128 offset:imm.
__device__ __forceinline__ v16bf loadA(const unsigned short* hA, int eoff) {
  uint4 t0 = *reinterpret_cast<const uint4*>(hA + eoff);
  uint4 t1 = *reinterpret_cast<const uint4*>(hA + eoff + 8);
  uint4 t[2] = { t0, t1 };
  v16bf v;
  __builtin_memcpy(&v, t, sizeof(v));
  return v;
}

// A fragment for the encoder input x (K padded 6 -> 32, zeros elsewhere).
// Only lanes 0-15 carry data (K 0..5); lanes 16-31 (K 16..31) are zero.
__device__ __forceinline__ v16bf loadAx(const float* __restrict__ xrow, int t, int lane) {
  unsigned short s[16];
#pragma unroll
  for (int i = 0; i < 16; ++i) s[i] = 0;
  if (lane < 16) {
    const float* p = xrow + (size_t)t * I_DIM;
    float2 a = *reinterpret_cast<const float2*>(p);
    float2 b = *reinterpret_cast<const float2*>(p + 2);
    float2 c = *reinterpret_cast<const float2*>(p + 4);
    s[0] = f2bf(a.x); s[1] = f2bf(a.y);
    s[2] = f2bf(b.x); s[3] = f2bf(b.y);
    s[4] = f2bf(c.x); s[5] = f2bf(c.y);
  }
  v16bf v;
  __builtin_memcpy(&v, s, sizeof(v));
  return v;
}

__device__ __forceinline__ void load_bias(float b[4][4], const float* __restrict__ bi,
                                          const float* __restrict__ bh, int ctbase, int n) {
#pragma unroll
  for (int ct = 0; ct < 4; ++ct)
#pragma unroll
    for (int q = 0; q < 4; ++q) {
      const int g = q * H_DIM + (ctbase + ct) * 16 + n;
      b[ct][q] = bi[g] + bh[g];
    }
}

// One LSTM cell step for this wave's 4 column tiles (64 hidden units).
// WinB/WhhB are lane-adjusted weight bases; houtB is the lane-adjusted LDS
// store base (hb + mhi*LSTR + ctbase*16 + n). All loop displacements are
// compile-time constants. kt-outer / q-inner: 4 independent WMMA chains.
template <int NKIN, int WINCOLS>
__device__ __forceinline__ void lstm_layer(const v16bf (&ain)[NKIN], const v16bf (&ah)[4],
                                           const unsigned short* __restrict__ WinB,
                                           const unsigned short* __restrict__ WhhB,
                                           const float (&bias)[4][4],
                                           v8f (&cst)[4],
                                           unsigned short* houtB) {
#pragma unroll
  for (int ct = 0; ct < 4; ++ct) {
    v8f acc[4];
    // first input K-tile: initialize the four gate chains from C = inline 0
#pragma unroll
    for (int q = 0; q < 4; ++q)
      acc[q] = __builtin_amdgcn_wmma_f32_16x16x32_bf16(
                 false, ain[0], false,
                 loadB(WinB, (q * H_DIM + ct * 16) * WINCOLS),
                 (short)0, (v8f){0, 0, 0, 0, 0, 0, 0, 0}, false, false);
#pragma unroll
    for (int kt = 1; kt < NKIN; ++kt)
#pragma unroll
      for (int q = 0; q < 4; ++q)
        acc[q] = __builtin_amdgcn_wmma_f32_16x16x32_bf16(
                   false, ain[kt], false,
                   loadB(WinB, (q * H_DIM + ct * 16) * WINCOLS + kt * 32),
                   (short)0, acc[q], false, false);
#pragma unroll
    for (int kt = 0; kt < 4; ++kt)
#pragma unroll
      for (int q = 0; q < 4; ++q)
        acc[q] = __builtin_amdgcn_wmma_f32_16x16x32_bf16(
                   false, ah[kt], false,
                   loadB(WhhB, (q * H_DIM + ct * 16) * H_DIM + kt * 32),
                   (short)0, acc[q], false, false);

    const float bi = bias[ct][0], bf_ = bias[ct][1];
    const float bg = bias[ct][2], bo  = bias[ct][3];
#pragma unroll
    for (int r = 0; r < 8; ++r) {
      const float iv = sigm(acc[0][r] + bi);
      const float fv = sigm(acc[1][r] + bf_);
      const float gv = tanh_(acc[2][r] + bg);
      const float ov = sigm(acc[3][r] + bo);
      const float cc = fv * cst[ct][r] + iv * gv;
      cst[ct][r] = cc;
      const float hv = ov * tanh_(cc);
      houtB[r * LSTR + ct * 16] = f2bf(hv);
    }
  }
}

struct Params {
  const float* __restrict__ x;
  const unsigned short* __restrict__ eWih0p;
  const unsigned short* __restrict__ eWhh0;
  const unsigned short* __restrict__ eWih1;
  const unsigned short* __restrict__ eWhh1;
  const unsigned short* __restrict__ dWih0;
  const unsigned short* __restrict__ dWhh0;
  const unsigned short* __restrict__ dWih1;
  const unsigned short* __restrict__ dWhh1;
  const float* __restrict__ eb_ih0;
  const float* __restrict__ eb_hh0;
  const float* __restrict__ eb_ih1;
  const float* __restrict__ eb_hh1;
  const float* __restrict__ db_ih0;
  const float* __restrict__ db_hh0;
  const float* __restrict__ db_ih1;
  const float* __restrict__ db_hh1;
  const float* __restrict__ fcW;
  const float* __restrict__ fcb;
  float* __restrict__ out;
};

__global__ __launch_bounds__(256) void lstm_s2s_kernel(Params p) {
  // h state: [layer][m-tile][16 rows][LSTR halves]  (bf16)
  __shared__ __align__(16) unsigned short hs[2][4][16 * LSTR];
  __shared__ float fcw_s[O_DIM * H_DIM];

  const int tid    = threadIdx.x;
  const int wave   = tid >> 5;
  const int lane   = tid & 31;
  const int mt     = wave >> 1;        // batch tile within block (0..3)
  const int half   = wave & 1;         // which 64 hidden cols this wave owns
  const int ctbase = half * 4;
  const int b0     = blockIdx.x * 64 + mt * 16;
  const int n      = lane & 15;
  const int m      = lane & 15;
  const int kh     = (lane >> 4) & 1;
  const int mhi    = (lane & 16) ? 8 : 0;

  for (int i = tid; i < 2 * 4 * 16 * LSTR; i += 256) (&hs[0][0][0])[i] = 0;
  fcw_s[tid] = p.fcW[tid];             // 256 threads load 2*128 floats

  // lane-adjusted weight base pointers: one per matrix, all inner-loop
  // displacements become instruction-immediate offsets.
  const int rowl = ctbase * 16 + n;    // lane's N-row within each gate block
  const unsigned short* eWih0pB = p.eWih0p + rowl * 32    + kh * 16;
  const unsigned short* eWhh0B  = p.eWhh0  + rowl * H_DIM + kh * 16;
  const unsigned short* eWih1B  = p.eWih1  + rowl * H_DIM + kh * 16;
  const unsigned short* eWhh1B  = p.eWhh1  + rowl * H_DIM + kh * 16;
  const unsigned short* dWih0B  = p.dWih0  + rowl * H_DIM + kh * 16;
  const unsigned short* dWhh0B  = p.dWhh0  + rowl * H_DIM + kh * 16;
  const unsigned short* dWih1B  = p.dWih1  + rowl * H_DIM + kh * 16;
  const unsigned short* dWhh1B  = p.dWhh1  + rowl * H_DIM + kh * 16;

  // lane-adjusted LDS bases for A-fragment loads and h stores
  unsigned short* h0b  = hs[0][mt];
  unsigned short* h1b  = hs[1][mt];
  const unsigned short* h0A = h0b + m * LSTR + kh * 16;
  const unsigned short* h1A = h1b + m * LSTR + kh * 16;
  unsigned short* h0oB = h0b + mhi * LSTR + ctbase * 16 + n;
  unsigned short* h1oB = h1b + mhi * LSTR + ctbase * 16 + n;

  const float* xrow = p.x + (size_t)(b0 + (lane & 15)) * (T_STEPS * I_DIM);

  v8f c0[4], c1[4];
#pragma unroll
  for (int ct = 0; ct < 4; ++ct) {
    c0[ct] = (v8f){0, 0, 0, 0, 0, 0, 0, 0};
    c1[ct] = (v8f){0, 0, 0, 0, 0, 0, 0, 0};
  }

  float bias0[4][4], bias1[4][4];
  load_bias(bias0, p.eb_ih0, p.eb_hh0, ctbase, n);
  load_bias(bias1, p.eb_ih1, p.eb_hh1, ctbase, n);
  __syncthreads();

  // ---------------- encoder: 100 timesteps ----------------
#pragma unroll 1
  for (int t = 0; t < T_STEPS; ++t) {
    v16bf ax[1];
    ax[0] = loadAx(xrow, t, lane);
    v16bf ah[4];
#pragma unroll
    for (int kt = 0; kt < 4; ++kt) ah[kt] = loadA(h0A, kt * 32);
    __syncthreads();   // all reads of h0_old done
    lstm_layer<1, 32>(ax, ah, eWih0pB, eWhh0B, bias0, c0, h0oB);
    __syncthreads();   // h0_new visible
    v16bf ai1[4], ah1[4];
#pragma unroll
    for (int kt = 0; kt < 4; ++kt) {
      ai1[kt] = loadA(h0A, kt * 32);
      ah1[kt] = loadA(h1A, kt * 32);
    }
    __syncthreads();   // all reads of h1_old done
    lstm_layer<4, H_DIM>(ai1, ah1, eWih1B, eWhh1B, bias1, c1, h1oB);
    __syncthreads();   // h1_new visible
  }

  // ---------------- decoder: 20 autoregressive steps ----------------
  load_bias(bias0, p.db_ih0, p.db_hh0, ctbase, n);
  load_bias(bias1, p.db_ih1, p.db_hh1, ctbase, n);

#pragma unroll 1
  for (int s = 0; s < OUTLEN; ++s) {
    v16bf ai0[4], ah0[4];
#pragma unroll
    for (int kt = 0; kt < 4; ++kt) {
      ai0[kt] = loadA(h1A, kt * 32);   // decoder input = previous h1
      ah0[kt] = loadA(h0A, kt * 32);
    }
    __syncthreads();
    lstm_layer<4, H_DIM>(ai0, ah0, dWih0B, dWhh0B, bias0, c0, h0oB);
    __syncthreads();
    v16bf ai1[4], ah1[4];
#pragma unroll
    for (int kt = 0; kt < 4; ++kt) {
      ai1[kt] = loadA(h0A, kt * 32);
      ah1[kt] = loadA(h1A, kt * 32);
    }
    __syncthreads();
    lstm_layer<4, H_DIM>(ai1, ah1, dWih1B, dWhh1B, bias1, c1, h1oB);
    __syncthreads();

    // FC head: pred[b, o] = h1[b,:] @ fcW[o,:] + fcb[o]  (O=2, VALU, fcW in LDS)
    if (tid < 128) {
      const int fmt = tid >> 5;
      const int rem = tid & 31;
      const int fm  = rem >> 1;
      const int o   = rem & 1;
      const unsigned short* hrow = hs[1][fmt] + fm * LSTR;
      float acc = p.fcb[o];
#pragma unroll 4
      for (int k = 0; k < H_DIM; ++k) acc += bf2f(hrow[k]) * fcw_s[o * H_DIM + k];
      p.out[(size_t)(blockIdx.x * 64 + fmt * 16 + fm) * (OUTLEN * O_DIM)
            + s * O_DIM + o] = acc;
    }
  }
}

// f32 -> bf16 conversion with optional column zero-padding (row-major).
__global__ void cvt_bf16_pad(unsigned short* __restrict__ dst, const float* __restrict__ src,
                             int rows, int scols, int dcols) {
  const int idx = blockIdx.x * blockDim.x + threadIdx.x;
  if (idx >= rows * dcols) return;
  const int r = idx / dcols;
  const int c = idx - r * dcols;
  const float v = (c < scols) ? src[(size_t)r * scols + c] : 0.f;
  dst[idx] = f2bf(v);
}

extern "C" void kernel_launch(void* const* d_in, const int* in_sizes, int n_in,
                              void* d_out, int out_size, void* d_ws, size_t ws_size,
                              hipStream_t stream) {
  (void)in_sizes; (void)n_in; (void)out_size; (void)ws_size;
  const float* x      = (const float*)d_in[0];
  const float* eW_ih0 = (const float*)d_in[1];
  const float* eW_hh0 = (const float*)d_in[2];
  const float* eb_ih0 = (const float*)d_in[3];
  const float* eb_hh0 = (const float*)d_in[4];
  const float* eW_ih1 = (const float*)d_in[5];
  const float* eW_hh1 = (const float*)d_in[6];
  const float* eb_ih1 = (const float*)d_in[7];
  const float* eb_hh1 = (const float*)d_in[8];
  const float* dW_ih0 = (const float*)d_in[9];
  const float* dW_hh0 = (const float*)d_in[10];
  const float* db_ih0 = (const float*)d_in[11];
  const float* db_hh0 = (const float*)d_in[12];
  const float* dW_ih1 = (const float*)d_in[13];
  const float* dW_hh1 = (const float*)d_in[14];
  const float* db_ih1 = (const float*)d_in[15];
  const float* db_hh1 = (const float*)d_in[16];
  const float* fc_W   = (const float*)d_in[17];
  const float* fc_b   = (const float*)d_in[18];

  // bf16 weight copies in workspace (~0.93 MB total, L2-resident).
  unsigned short* w      = (unsigned short*)d_ws;
  unsigned short* eWih0p = w;                       // 512 x 32 (K padded 6->32)
  unsigned short* eWhh0  = eWih0p + 512 * 32;       // 512 x 128 each below
  unsigned short* eWih1  = eWhh0  + 512 * 128;
  unsigned short* eWhh1  = eWih1  + 512 * 128;
  unsigned short* dWih0  = eWhh1  + 512 * 128;
  unsigned short* dWhh0  = dWih0  + 512 * 128;
  unsigned short* dWih1  = dWhh0  + 512 * 128;
  unsigned short* dWhh1  = dWih1  + 512 * 128;

  const int thr = 256;
  const int nPad  = 512 * 32;
  const int nFull = 512 * 128;
  cvt_bf16_pad<<<(nPad  + thr - 1) / thr, thr, 0, stream>>>(eWih0p, eW_ih0, 512, 6,   32);
  cvt_bf16_pad<<<(nFull + thr - 1) / thr, thr, 0, stream>>>(eWhh0,  eW_hh0, 512, 128, 128);
  cvt_bf16_pad<<<(nFull + thr - 1) / thr, thr, 0, stream>>>(eWih1,  eW_ih1, 512, 128, 128);
  cvt_bf16_pad<<<(nFull + thr - 1) / thr, thr, 0, stream>>>(eWhh1,  eW_hh1, 512, 128, 128);
  cvt_bf16_pad<<<(nFull + thr - 1) / thr, thr, 0, stream>>>(dWih0,  dW_ih0, 512, 128, 128);
  cvt_bf16_pad<<<(nFull + thr - 1) / thr, thr, 0, stream>>>(dWhh0,  dW_hh0, 512, 128, 128);
  cvt_bf16_pad<<<(nFull + thr - 1) / thr, thr, 0, stream>>>(dWih1,  dW_ih1, 512, 128, 128);
  cvt_bf16_pad<<<(nFull + thr - 1) / thr, thr, 0, stream>>>(dWhh1,  dW_hh1, 512, 128, 128);

  Params p{ x, eWih0p, eWhh0, eWih1, eWhh1, dWih0, dWhh0, dWih1, dWhh1,
            eb_ih0, eb_hh0, eb_ih1, eb_hh1, db_ih0, db_hh0, db_ih1, db_hh1,
            fc_W, fc_b, (float*)d_out };

  lstm_s2s_kernel<<<16384 / 64, 256, 0, stream>>>(p);
}
